// dense_conv_55791625175545
// MI455X (gfx1250) — matrix operand order repaired and checked by reading, not compiled, not run on hardware
//
#include <hip/hip_runtime.h>
#include <hip/hip_bf16.h>

// ---------------------------------------------------------------------------
// DGCNN EdgeConv block for MI455X (gfx1250, wave32, WMMA bf16 16x16x32)
// B=4, N=4096, C=24, G=64, K=16
// ---------------------------------------------------------------------------

#define BATCH 4
#define NPTS  4096
#define CH    24
#define GDIM  64
#define KNN   16

typedef __attribute__((ext_vector_type(16))) __bf16 v16bf;
typedef __attribute__((ext_vector_type(8)))  __bf16 v8bf;
typedef __attribute__((ext_vector_type(8)))  float  v8f;

union BF16x16 { v16bf v; v8bf h[2]; };

__device__ __forceinline__ __bf16 f2bf(float f) {
    union { float f; unsigned u; } v; v.f = f;
    unsigned r = v.u + 0x7FFFu + ((v.u >> 16) & 1u);   // round-to-nearest-even
    unsigned short hs = (unsigned short)(r >> 16);
    return __builtin_bit_cast(__bf16, hs);
}

// ---------------------------------------------------------------------------
// Kernel 1: prep
//  - points (f32 [B,N,24]) -> bf16 padded [B*N][32] (K-pad to 32) + f32 norms
//  - W1 (48x64) -> W1t [g][64]  (K-pad 48->64), bf16, transposed
//  - W3 (88x64) -> W3t [g][96]  (K-pad 88->96)
//  - W2 (152x64)-> W2t [g][160] rows PERMUTED to match LDS feature layout:
//       cols 0..63  <- W2 rows 64..127  (y1)
//       cols 64..87 <- W2 rows 128..151 (central)
//       cols 88..95 <- 0                (pad)
//       cols 96..159<- W2 rows 0..63    (z)
// ---------------------------------------------------------------------------
__global__ __launch_bounds__(256)
void prep_kernel(const float* __restrict__ pts,
                 const float* __restrict__ W1,
                 const float* __restrict__ W2,
                 const float* __restrict__ W3,
                 __bf16* __restrict__ ptsb, float* __restrict__ sq,
                 __bf16* __restrict__ W1t, __bf16* __restrict__ W3t,
                 __bf16* __restrict__ W2t) {
    int gid = blockIdx.x;
    if (gid < (BATCH * NPTS) / 256) {
        int p = gid * 256 + threadIdx.x;
        const float* src = pts + (size_t)p * CH;
        __bf16* dst = ptsb + (size_t)p * 32;
        float s = 0.f;
        #pragma unroll
        for (int c = 0; c < CH; ++c) { float v = src[c]; s += v * v; dst[c] = f2bf(v); }
        #pragma unroll
        for (int c = CH; c < 32; ++c) dst[c] = f2bf(0.f);
        sq[p] = s;
    } else {
        for (int e = threadIdx.x; e < GDIM * 64; e += 256) {
            int g = e >> 6, k = e & 63;
            W1t[g * 64 + k] = f2bf(k < 48 ? W1[k * GDIM + g] : 0.f);
        }
        for (int e = threadIdx.x; e < GDIM * 96; e += 256) {
            int g = e / 96, k = e % 96;
            W3t[g * 96 + k] = f2bf(k < 88 ? W3[k * GDIM + g] : 0.f);
        }
        for (int e = threadIdx.x; e < GDIM * 160; e += 256) {
            int g = e / 160, c = e % 160;
            float v;
            if      (c < 64) v = W2[(64 + c) * GDIM + g];
            else if (c < 88) v = W2[(128 + (c - 64)) * GDIM + g];
            else if (c < 96) v = 0.f;
            else             v = W2[(c - 96) * GDIM + g];
            W2t[g * 160 + c] = f2bf(v);
        }
    }
}

// ---------------------------------------------------------------------------
// Kernel 2: KNN via WMMA Gram tiles.
//  One wave per 16-row tile. Metric per row m: d(n) = |p_n|^2 - 2 <p_m, p_n>
//  (row norm constant per row -> dropped; ordering unchanged).
//  Each 16x16 tile = ONE v_wmma_f32_16x16x32_bf16 (K=32 covers 24 channels).
//  2 lanes per row keep private sorted top-16 lists; merged at the end.
// ---------------------------------------------------------------------------
__global__ __launch_bounds__(32)
void knn_kernel(const __bf16* __restrict__ ptsb,
                const float* __restrict__ sq,
                int* __restrict__ idx_out,
                float* __restrict__ idx_f_out) {
    __shared__ __align__(16) float dtile[16][16];
    __shared__ float ldist[2][16][16];
    __shared__ int   lidx [2][16][16];

    int tile = blockIdx.x;          // B * N/16 tiles
    int b    = tile >> 8;           // 256 row-tiles per batch
    int rt   = tile & 255;
    int lane = threadIdx.x;
    int m    = lane & 15;
    int h    = lane >> 4;
    int rowBase = b * NPTS + rt * 16;

    // A tile (16x32 bf16): lane<16 holds K 0..7 & 16..23 of row m; lane>=16: 8..15 & 24..31
    const __bf16* arow = ptsb + (size_t)(rowBase + m) * 32;
    BF16x16 A;
    A.h[0] = *(const v8bf*)(arow + h * 8);
    A.h[1] = *(const v8bf*)(arow + 16 + h * 8);

    float bd[KNN]; int bi[KNN];
    #pragma unroll
    for (int j = 0; j < KNN; ++j) { bd[j] = 3.0e38f; bi[j] = 0; }

    for (int ct = 0; ct < NPTS / 16; ++ct) {
        // B tile (32x16 bf16): lane's column = ct*16 + m, K 8h..8h+7 & 16+8h..16+8h+7
        const __bf16* bcol = ptsb + (size_t)(b * NPTS + ct * 16 + m) * 32 + h * 8;
        BF16x16 Bm;
        Bm.h[0] = *(const v8bf*)(bcol);
        Bm.h[1] = *(const v8bf*)(bcol + 16);

        v8f acc = {};
        acc = __builtin_amdgcn_wmma_f32_16x16x32_bf16(false, A.v, false, Bm.v,
                                                      (short)0, acc, false, false);

        float sqc = sq[b * NPTS + ct * 16 + m];   // this lane's column norm
        #pragma unroll
        for (int r = 0; r < 8; ++r)
            dtile[r + h * 8][m] = sqc - 2.0f * acc[r];
        __syncthreads();

        // lane handles row m, columns n = 2t + h
        #pragma unroll
        for (int t = 0; t < 8; ++t) {
            int n = 2 * t + h;
            float d = dtile[m][n];
            int c = ct * 16 + n;
            if (c != rt * 16 + m && d < bd[KNN - 1]) {
                bd[KNN - 1] = d; bi[KNN - 1] = c;
                #pragma unroll
                for (int j = KNN - 1; j > 0; --j) {
                    if (bd[j] < bd[j - 1]) {
                        float td = bd[j]; bd[j] = bd[j - 1]; bd[j - 1] = td;
                        int   ti = bi[j]; bi[j] = bi[j - 1]; bi[j - 1] = ti;
                    }
                }
            }
        }
        __syncthreads();
    }

    #pragma unroll
    for (int j = 0; j < KNN; ++j) { ldist[h][m][j] = bd[j]; lidx[h][m][j] = bi[j]; }
    __syncthreads();

    if (lane < 16) {
        int i0 = 0, i1 = 0;
        size_t base = (size_t)(rowBase + m) * KNN;
        for (int j = 0; j < KNN; ++j) {
            float d0 = ldist[0][m][i0], d1 = ldist[1][m][i1];
            bool p = d1 < d0;
            int id = p ? lidx[1][m][i1] : lidx[0][m][i0];
            if (p) ++i1; else ++i0;
            idx_out[base + j]   = id;
            idx_f_out[base + j] = (float)id;
        }
    }
}

// ---------------------------------------------------------------------------
// Kernel 3: edge-MLP + max-pool. One point per wave, 4 waves/block.
// LDS feature strip per wave: 16 rows x 160 bf16:
//   after pass A : [edge 0..47 | pad 48..63 | central 64..87 | pad 88..95 | - ]
//   after stage1 : [y1   0..63 | central 64..87 | pad | -              ]
//   after stage2 : [y1 | central | pad | z 96..159 ]
// Stage GEMMs: 16x64 result, kt steps of K=32, 4 n-tiles of 16.
// ---------------------------------------------------------------------------
__global__ __launch_bounds__(128)
void mlp_kernel(const float* __restrict__ pts,
                const int* __restrict__ idx,
                const __bf16* __restrict__ W1t,
                const __bf16* __restrict__ W3t,
                const __bf16* __restrict__ W2t,
                const float* __restrict__ b1,
                const float* __restrict__ b3,
                const float* __restrict__ b2,
                float* __restrict__ out) {
    __shared__ __align__(16) __bf16 feat[4][16][160];

    int wv   = threadIdx.x >> 5;
    int lane = threadIdx.x & 31;
    int m    = lane & 15;
    int h    = lane >> 4;
    int p    = blockIdx.x * 4 + wv;        // global point id
    int b    = p >> 12;                    // / NPTS

    __bf16 (*F)[160] = feat[wv];
    const float* central = pts + (size_t)p * CH;
    int nb = idx[(size_t)p * KNN + m];
    const float* nbr = pts + (size_t)(b * NPTS + nb) * CH;

    // ---- pass A: edge features + central copy + zero pads
    #pragma unroll
    for (int cc = 0; cc < 12; ++cc) {
        int c = h * 12 + cc;
        float ce = central[c];
        F[m][c]        = f2bf(ce);
        F[m][24 + c]   = f2bf(nbr[c] - ce);
        F[m][64 + c]   = f2bf(ce);
    }
    if (h == 0) {
        #pragma unroll
        for (int c = 48; c < 56; ++c) F[m][c] = f2bf(0.f);
        #pragma unroll
        for (int c = 88; c < 92; ++c) F[m][c] = f2bf(0.f);
    } else {
        #pragma unroll
        for (int c = 56; c < 64; ++c) F[m][c] = f2bf(0.f);
        #pragma unroll
        for (int c = 92; c < 96; ++c) F[m][c] = f2bf(0.f);
    }

    v8f acc[4];

    // ---- stage 1: edge(48->64) @ W1 -> y1, relu, write to cols 0..63
    #pragma unroll
    for (int nt = 0; nt < 4; ++nt) acc[nt] = (v8f){};
    #pragma unroll
    for (int kt = 0; kt < 2; ++kt) {
        BF16x16 a;
        a.h[0] = *(const v8bf*)&F[m][kt * 32 + h * 8];
        a.h[1] = *(const v8bf*)&F[m][kt * 32 + 16 + h * 8];
        #pragma unroll
        for (int nt = 0; nt < 4; ++nt) {
            const __bf16* wp = W1t + (size_t)(nt * 16 + m) * 64 + kt * 32 + h * 8;
            BF16x16 bm; bm.h[0] = *(const v8bf*)wp; bm.h[1] = *(const v8bf*)(wp + 16);
            acc[nt] = __builtin_amdgcn_wmma_f32_16x16x32_bf16(false, a.v, false, bm.v,
                                                              (short)0, acc[nt], false, false);
        }
    }
    #pragma unroll
    for (int nt = 0; nt < 4; ++nt) {
        int col = nt * 16 + m;
        float bias = b1[col];
        #pragma unroll
        for (int r = 0; r < 8; ++r) {
            float v = acc[nt][r] + bias; v = v > 0.f ? v : 0.f;
            F[r + h * 8][col] = f2bf(v);
        }
    }

    // ---- stage 2: [y1|central](88->96) @ W3 -> z, relu, write to cols 96..159
    #pragma unroll
    for (int nt = 0; nt < 4; ++nt) acc[nt] = (v8f){};
    #pragma unroll
    for (int kt = 0; kt < 3; ++kt) {
        BF16x16 a;
        a.h[0] = *(const v8bf*)&F[m][kt * 32 + h * 8];
        a.h[1] = *(const v8bf*)&F[m][kt * 32 + 16 + h * 8];
        #pragma unroll
        for (int nt = 0; nt < 4; ++nt) {
            const __bf16* wp = W3t + (size_t)(nt * 16 + m) * 96 + kt * 32 + h * 8;
            BF16x16 bm; bm.h[0] = *(const v8bf*)wp; bm.h[1] = *(const v8bf*)(wp + 16);
            acc[nt] = __builtin_amdgcn_wmma_f32_16x16x32_bf16(false, a.v, false, bm.v,
                                                              (short)0, acc[nt], false, false);
        }
    }
    #pragma unroll
    for (int nt = 0; nt < 4; ++nt) {
        int col = nt * 16 + m;
        float bias = b3[col];
        #pragma unroll
        for (int r = 0; r < 8; ++r) {
            float v = acc[nt][r] + bias; v = v > 0.f ? v : 0.f;
            F[r + h * 8][96 + col] = f2bf(v);
        }
    }

    // ---- stage 3: [y1|central|pad|z](152->160) @ permuted W2 -> w (no relu)
    #pragma unroll
    for (int nt = 0; nt < 4; ++nt) acc[nt] = (v8f){};
    #pragma unroll
    for (int kt = 0; kt < 5; ++kt) {
        BF16x16 a;
        a.h[0] = *(const v8bf*)&F[m][kt * 32 + h * 8];
        a.h[1] = *(const v8bf*)&F[m][kt * 32 + 16 + h * 8];
        #pragma unroll
        for (int nt = 0; nt < 4; ++nt) {
            const __bf16* wp = W2t + (size_t)(nt * 16 + m) * 160 + kt * 32 + h * 8;
            BF16x16 bm; bm.h[0] = *(const v8bf*)wp; bm.h[1] = *(const v8bf*)(wp + 16);
            acc[nt] = __builtin_amdgcn_wmma_f32_16x16x32_bf16(false, a.v, false, bm.v,
                                                              (short)0, acc[nt], false, false);
        }
    }

    // ---- max over k, output order: [w(64) | z(64) | y1(64) | central(24)]
    float* orow = out + (size_t)p * 216;
    #pragma unroll
    for (int nt = 0; nt < 4; ++nt) {
        int col = nt * 16 + m;
        float bias = b2[col];
        float pm = -3.0e38f;
        #pragma unroll
        for (int r = 0; r < 8; ++r) { float v = acc[nt][r] + bias; pm = pm > v ? pm : v; }
        float om = __shfl_xor(pm, 16, 32);   // rows 0..7 live in lane, 8..15 in lane^16
        pm = pm > om ? pm : om;
        if (h == 0) orow[col] = pm;
    }
    for (int j = lane; j < 64; j += 32) {
        float mz = -3.0e38f, my = -3.0e38f;
        #pragma unroll
        for (int r = 0; r < 16; ++r) {
            float vz = (float)F[r][96 + j]; mz = mz > vz ? mz : vz;
            float vy = (float)F[r][j];      my = my > vy ? my : vy;
        }
        orow[64 + j]  = mz;
        orow[128 + j] = my;
    }
    for (int j = lane; j < CH; j += 32) orow[192 + j] = central[j];
}

// ---------------------------------------------------------------------------
extern "C" void kernel_launch(void* const* d_in, const int* in_sizes, int n_in,
                              void* d_out, int out_size, void* d_ws, size_t ws_size,
                              hipStream_t stream) {
    const float* pts = (const float*)d_in[0];
    const float* W1  = (const float*)d_in[1];
    const float* b1  = (const float*)d_in[2];
    const float* W2  = (const float*)d_in[3];
    const float* b2  = (const float*)d_in[4];
    const float* W3  = (const float*)d_in[5];
    const float* b3  = (const float*)d_in[6];

    char* ws = (char*)d_ws;
    __bf16* ptsb  = (__bf16*)(ws);                               // 16384*32*2 = 1048576
    float*  sq    = (float*) (ws + 1048576);                     // 65536
    int*    idxb  = (int*)   (ws + 1048576 + 65536);             // 1048576
    __bf16* W1t   = (__bf16*)(ws + 2162688);                     // 8192
    __bf16* W3t   = (__bf16*)(ws + 2170880);                     // 12288
    __bf16* W2t   = (__bf16*)(ws + 2183168);                     // 20480

    float* out   = (float*)d_out;                                // [B,N,216]
    float* idx_f = out + (size_t)BATCH * NPTS * 216;             // [B,N,16] as float

    prep_kernel<<<(BATCH * NPTS) / 256 + 1, 256, 0, stream>>>(
        pts, W1, W2, W3, ptsb, sq, W1t, W3t, W2t);

    knn_kernel<<<BATCH * (NPTS / 16), 32, 0, stream>>>(ptsb, sq, idxb, idx_f);

    mlp_kernel<<<(BATCH * NPTS) / 4, 128, 0, stream>>>(
        pts, idxb, W1t, W3t, W2t, b1, b3, b2, out);
}
